// Level_28209345200495
// MI455X (gfx1250) — matrix operand order repaired and checked by reading, not compiled, not run on hardware
//
#include <hip/hip_runtime.h>

// ---------------------------------------------------------------------------
// Problem constants (from reference setup_inputs)
// ---------------------------------------------------------------------------
#define BB 4          // batch
#define NP 4096       // points per batch
#define MP 1024       // previous points
#define CF 264        // final feature channels
#define KNN_K 16
#define FMK 5
#define RT 4          // RATIO
#define COLS (BB*NP*RT)   // 65536 GEMM columns
#define KP1 288           // padded K for up1 (266 -> 288, mult of 32)

typedef __attribute__((ext_vector_type(16))) _Float16 v16h;
typedef __attribute__((ext_vector_type(2)))  _Float16 v2h;
typedef __attribute__((ext_vector_type(8)))  float    v8f;
typedef __attribute__((ext_vector_type(2)))  float    v2f;

// ---------------------------------------------------------------------------
// K0: per-batch centroid + radius for normalize_point_batch
// ---------------------------------------------------------------------------
__global__ void stats_kernel(const float* __restrict__ xyz, float* __restrict__ stats) {
    int b = blockIdx.x, tid = threadIdx.x;
    __shared__ float red[256];
    __shared__ float cen[3];
    const float* xb = xyz + (size_t)b * 3 * NP;
    float s[3] = {0.f, 0.f, 0.f};
    for (int n = tid; n < NP; n += 256) {
        s[0] += xb[n]; s[1] += xb[NP + n]; s[2] += xb[2 * NP + n];
    }
    for (int c = 0; c < 3; ++c) {
        red[tid] = s[c]; __syncthreads();
        for (int st = 128; st > 0; st >>= 1) {
            if (tid < st) red[tid] += red[tid + st];
            __syncthreads();
        }
        if (tid == 0) cen[c] = red[0] * (1.0f / NP);
        __syncthreads();
    }
    float mx = 0.f;
    for (int n = tid; n < NP; n += 256) {
        float dx = xb[n] - cen[0], dy = xb[NP + n] - cen[1], dz = xb[2 * NP + n] - cen[2];
        mx = fmaxf(mx, sqrtf(dx * dx + dy * dy + dz * dz));
    }
    red[tid] = mx; __syncthreads();
    for (int st = 128; st > 0; st >>= 1) {
        if (tid < st) red[tid] = fmaxf(red[tid], red[tid + st]);
        __syncthreads();
    }
    if (tid == 0) {
        stats[b * 4 + 0] = cen[0]; stats[b * 4 + 1] = cen[1];
        stats[b * 4 + 2] = cen[2]; stats[b * 4 + 3] = red[0];
    }
}

// ---------------------------------------------------------------------------
// K1: x_norm + l0 conv1d (3->24). Writes feat24 and F channels [240..263].
// ---------------------------------------------------------------------------
__global__ void l0_kernel(const float* __restrict__ xyz, const float* __restrict__ stats,
                          const float* __restrict__ W, const float* __restrict__ bias,
                          float* __restrict__ feat24, float* __restrict__ F) {
    int p = blockIdx.x * 256 + threadIdx.x;   // global point id
    int b = p >> 12, n = p & (NP - 1);
    const float* st = stats + b * 4;
    float inv = 1.0f / st[3];
    float xn[3];
#pragma unroll
    for (int c = 0; c < 3; ++c)
        xn[c] = (xyz[((size_t)b * 3 + c) * NP + n] - st[c]) * inv;
    float* f24 = feat24 + (size_t)p * 24;
    float* Fo  = F + (size_t)p * CF + 240;
#pragma unroll
    for (int o = 0; o < 24; ++o) {
        float s2 = bias[o] + W[o * 3] * xn[0] + W[o * 3 + 1] * xn[1] + W[o * 3 + 2] * xn[2];
        f24[o] = s2; Fo[o] = s2;
    }
}

// ---------------------------------------------------------------------------
// K2: self-KNN over 24-dim features via f32 WMMA cross products.
// Ranking metric: |c|^2 - 2*q.c  (|q|^2 constant per query -> dropped).
// Block: 256 threads = 8 waves = 256 queries (16 query tiles; 2 per wave).
// Per 256-candidate chunk staged in LDS: each wave runs 16 candidate tiles,
// 6 x v_wmma_f32_16x16x4_f32 per (query-tile, cand-tile), spills the 16x16
// score tile to wave-private LDS, then each lane inserts its query's 16
// candidate scores into its private top-16 list.
// ---------------------------------------------------------------------------
__global__ void knn_wmma_kernel(const float* __restrict__ feat24, int* __restrict__ knn) {
    __shared__ float chF[256 * 24];     // candidate chunk features
    __shared__ float chN[256];          // candidate norms
    __shared__ float scores[8][2][256]; // [wave][qtile][M*16+N] wave-private
    int tid = threadIdx.x, lane = tid & 31, wave = tid >> 5;
    const int bpb = NP / 256;
    int b = blockIdx.x / bpb;
    int q0blk = (blockIdx.x % bpb) * 256;
    const float* fb = feat24 + (size_t)b * NP * 24;
    int nOwn = q0blk + wave * 32 + lane;      // query owned by this lane
    int nloc = lane & 15;
    int akb  = (lane & 16) ? 2 : 0;           // K base for A/B fragments

    // A fragments for this wave's 2 query tiles (lane -> M = lane%16)
    v2f afr[2][6];
#pragma unroll
    for (int t = 0; t < 2; ++t) {
        const float* ap = fb + (size_t)(q0blk + wave * 32 + t * 16 + nloc) * 24 + akb;
#pragma unroll
        for (int s = 0; s < 6; ++s) afr[t][s] = *(const v2f*)(ap + 4 * s);
    }

    float dk[KNN_K]; int ik[KNN_K];
#pragma unroll
    for (int t = 0; t < KNN_K; ++t) { dk[t] = 1e30f; ik[t] = 0; }

    for (int c0 = 0; c0 < NP; c0 += 256) {
        __syncthreads();
        for (int i = tid; i < 256 * 24; i += 256) chF[i] = fb[(size_t)c0 * 24 + i];
        if (c0 + 256 < NP)                         // gfx1250 global_prefetch_b8
            __builtin_prefetch(fb + (size_t)(c0 + 256) * 24 + (size_t)tid * 24, 0, 0);
        __syncthreads();
        {   // candidate norms
            const float* cf = chF + tid * 24;
            float s = 0.f;
#pragma unroll
            for (int c = 0; c < 24; ++c) s += cf[c] * cf[c];
            chN[tid] = s;
        }
        __syncthreads();

        for (int ct = 0; ct < 16; ++ct) {
            const float* bp = chF + (ct * 16 + nloc) * 24 + akb;   // lane -> N
            float cn = chN[ct * 16 + nloc];
            v2f bfr[6];
#pragma unroll
            for (int s = 0; s < 6; ++s) bfr[s] = *(const v2f*)(bp + 4 * s);
            int mb = (lane & 16) ? 8 : 0;
#pragma unroll
            for (int t = 0; t < 2; ++t) {
                v8f acc = {};
#pragma unroll
                for (int s = 0; s < 6; ++s)
                    acc = __builtin_amdgcn_wmma_f32_16x16x4_f32(
                        false, afr[t][s], false, bfr[s], (short)0, acc, false, false);
#pragma unroll
                for (int v = 0; v < 8; ++v)
                    scores[wave][t][(mb + v) * 16 + nloc] = cn - 2.f * acc[v];
            }
            // same-wave LDS ops are in-order: safe to read scores back now
            int myT = lane >> 4, myQ = lane & 15;
            const float* sr = &scores[wave][myT][myQ * 16];
#pragma unroll
            for (int cc = 0; cc < 16; ++cc) {
                int j = c0 + ct * 16 + cc;
                float sc = sr[cc];
                if (j != nOwn && sc < dk[KNN_K - 1]) {
                    float dc = sc; int ic = j;
#pragma unroll
                    for (int t = 0; t < KNN_K; ++t) {
                        if (dc < dk[t]) {
                            float td = dk[t]; int ti = ik[t];
                            dk[t] = dc; ik[t] = ic; dc = td; ic = ti;
                        }
                    }
                }
            }
        }
    }
    int* op = knn + ((size_t)b * NP + nOwn) * KNN_K;
#pragma unroll
    for (int t = 0; t < KNN_K; ++t) op[t] = ik[t];
}

// ---------------------------------------------------------------------------
// K3: edge MLP + max-pool over 16 neighbors. Writes 60 channels at chOff:
// [max h2 (12), max h1 (12), max h0 (12), center (24)]
// ---------------------------------------------------------------------------
__global__ void edge_kernel(const float* __restrict__ feat24, const int* __restrict__ knn,
                            const float* __restrict__ W0, const float* __restrict__ b0,
                            const float* __restrict__ W1, const float* __restrict__ b1,
                            const float* __restrict__ W2, const float* __restrict__ b2,
                            float* __restrict__ F, int chOff) {
    __shared__ float sw[1620];
    int tid = threadIdx.x;
    for (int i = tid; i < 576; i += 128) sw[i] = W0[i];
    for (int i = tid; i < 432; i += 128) sw[576 + i] = W1[i];
    for (int i = tid; i < 576; i += 128) sw[1008 + i] = W2[i];
    if (tid < 12) { sw[1584 + tid] = b0[tid]; sw[1596 + tid] = b1[tid]; sw[1608 + tid] = b2[tid]; }
    __syncthreads();
    int p = blockIdx.x * 128 + tid;
    int b = p >> 12;
    int bbase = b << 12;
    float q[24];
#pragma unroll
    for (int c = 0; c < 24; ++c) q[c] = feat24[(size_t)p * 24 + c];
    float mh[36];
#pragma unroll
    for (int c = 0; c < 36; ++c) mh[c] = -1e30f;
    for (int kn = 0; kn < KNN_K; ++kn) {
        int j = knn[(size_t)p * KNN_K + kn];
        const float* nf = feat24 + (size_t)(bbase + j) * 24;
        float e[24];
#pragma unroll
        for (int c = 0; c < 24; ++c) e[c] = nf[c] - q[c];
        float h0[12], h1[12], h2[12];
#pragma unroll
        for (int o = 0; o < 12; ++o) {            // W0: 12 x 48 over [q, nb-q]
            const float* wr = &sw[o * 48];
            float s2 = sw[1584 + o];
#pragma unroll
            for (int c = 0; c < 24; ++c) s2 += wr[c] * q[c];
#pragma unroll
            for (int c = 0; c < 24; ++c) s2 += wr[24 + c] * e[c];
            h0[o] = fmaxf(s2, 0.f);
        }
#pragma unroll
        for (int o = 0; o < 12; ++o) {            // W1: 12 x 36 over [h0, q]
            const float* wr = &sw[576 + o * 36];
            float s2 = sw[1596 + o];
#pragma unroll
            for (int c = 0; c < 12; ++c) s2 += wr[c] * h0[c];
#pragma unroll
            for (int c = 0; c < 24; ++c) s2 += wr[12 + c] * q[c];
            h1[o] = fmaxf(s2, 0.f);
        }
#pragma unroll
        for (int o = 0; o < 12; ++o) {            // W2: 12 x 48 over [h1, h0, q]
            const float* wr = &sw[1008 + o * 48];
            float s2 = sw[1608 + o];
#pragma unroll
            for (int c = 0; c < 12; ++c) s2 += wr[c] * h1[c];
#pragma unroll
            for (int c = 0; c < 12; ++c) s2 += wr[12 + c] * h0[c];
#pragma unroll
            for (int c = 0; c < 24; ++c) s2 += wr[24 + c] * q[c];
            h2[o] = s2;                           // no relu on last layer
        }
#pragma unroll
        for (int o = 0; o < 12; ++o) {
            mh[o]      = fmaxf(mh[o], h2[o]);
            mh[12 + o] = fmaxf(mh[12 + o], h1[o]);
            mh[24 + o] = fmaxf(mh[24 + o], h0[o]);
        }
    }
    float* Fo = F + (size_t)p * CF + chOff;
#pragma unroll
    for (int c = 0; c < 36; ++c) Fo[c] = mh[c];
#pragma unroll
    for (int c = 0; c < 24; ++c) Fo[36 + c] = q[c];
}

// ---------------------------------------------------------------------------
// K4: pointwise conv (Cin -> 24) with relu; input = F channels [chOff..263]
// ---------------------------------------------------------------------------
__global__ void pconv_kernel(const float* __restrict__ F, const float* __restrict__ W,
                             const float* __restrict__ bias, float* __restrict__ out24,
                             int chOff, int Cin) {
    __shared__ float sw[24 * 204];
    __shared__ float sb[24];
    int tid = threadIdx.x;
    for (int i = tid; i < 24 * Cin; i += 256) sw[i] = W[i];
    if (tid < 24) sb[tid] = bias[tid];
    __syncthreads();
    int p = blockIdx.x * 256 + tid;
    const float* fin = F + (size_t)p * CF + chOff;
    float acc[24];
#pragma unroll
    for (int o = 0; o < 24; ++o) acc[o] = sb[o];
    for (int c = 0; c < Cin; ++c) {
        float fv = fin[c];
#pragma unroll
        for (int o = 0; o < 24; ++o) acc[o] += sw[o * Cin + c] * fv;
    }
    float* op = out24 + (size_t)p * 24;
#pragma unroll
    for (int o = 0; o < 24; ++o) op[o] = fmaxf(acc[o], 0.f);
}

// ---------------------------------------------------------------------------
// K5: feature-matching KNN (5-NN in xyz vs previous_xyz), plus per-block
// partial sums of spatial distances (for the exp_weight bandwidth h).
// ---------------------------------------------------------------------------
__global__ void fm_knn_kernel(const float* __restrict__ xyz, const float* __restrict__ pxyz,
                              int* __restrict__ idx, float* __restrict__ dsp,
                              float* __restrict__ psum) {
    __shared__ float sp[3 * MP];
    __shared__ float red[256];
    int tid = threadIdx.x;
    const int bpb = NP / 256;
    int b = blockIdx.x / bpb;
    int n = (blockIdx.x % bpb) * 256 + tid;
    for (int i = tid; i < 3 * MP; i += 256) sp[i] = pxyz[(size_t)b * 3 * MP + i];
    __syncthreads();
    float qx = xyz[((size_t)b * 3 + 0) * NP + n];
    float qy = xyz[((size_t)b * 3 + 1) * NP + n];
    float qz = xyz[((size_t)b * 3 + 2) * NP + n];
    float dk[FMK]; int ik[FMK];
#pragma unroll
    for (int t = 0; t < FMK; ++t) { dk[t] = 1e30f; ik[t] = 0; }
    for (int m = 0; m < MP; ++m) {
        float dx = qx - sp[m], dy = qy - sp[MP + m], dz = qz - sp[2 * MP + m];
        float d = dx * dx + dy * dy + dz * dz;
        if (d < dk[FMK - 1]) {
            float dc = d; int ic = m;
#pragma unroll
            for (int t = 0; t < FMK; ++t) {
                if (dc < dk[t]) {
                    float td = dk[t]; int ti = ik[t];
                    dk[t] = dc; ik[t] = ic; dc = td; ic = ti;
                }
            }
        }
    }
    size_t base = ((size_t)b * NP + n) * FMK;
    float ls = 0.f;
#pragma unroll
    for (int t = 0; t < FMK; ++t) {
        idx[base + t] = ik[t];
        float v = dk[t] * (1.0f / 3.0f);   // mean over 3 coords
        dsp[base + t] = v; ls += v;
    }
    red[tid] = ls; __syncthreads();
    for (int st = 128; st > 0; st >>= 1) {
        if (tid < st) red[tid] += red[tid + st];
        __syncthreads();
    }
    if (tid == 0) psum[blockIdx.x] = red[0];
}

// ---------------------------------------------------------------------------
// K6: feature-space distances to the 5 matched prev points + partial sums
// ---------------------------------------------------------------------------
__global__ void fm_feat_kernel(const float* __restrict__ F, const float* __restrict__ pfeat,
                               const int* __restrict__ idx, float* __restrict__ dft,
                               float* __restrict__ psum) {
    __shared__ float red[128];
    int tid = threadIdx.x;
    int p = blockIdx.x * 128 + tid;
    int b = p >> 12;
    const float* fq  = F + (size_t)p * CF;
    const float* pfb = pfeat + (size_t)b * CF * MP;
    size_t base = (size_t)p * FMK;
    float ls = 0.f;
    for (int k = 0; k < FMK; ++k) {
        int m = idx[base + k];
        float s = 0.f;
        for (int c = 0; c < CF; ++c) {
            float df = fq[c] - pfb[(size_t)c * MP + m];
            s += df * df;
        }
        float v = s * (1.0f / CF);
        dft[base + k] = v; ls += v;
    }
    red[tid] = ls; __syncthreads();
    for (int st = 64; st > 0; st >>= 1) {
        if (tid < st) red[tid] += red[tid + st];
        __syncthreads();
    }
    if (tid == 0) psum[blockIdx.x] = red[0];
}

// K7: deterministic fixed-order reduction of partial sums -> hacc[b*2+{0,1}]
__global__ void fm_reduce_kernel(const float* __restrict__ psp, const float* __restrict__ pft,
                                 float* __restrict__ hacc) {
    int id = threadIdx.x;
    if (id >= 2 * BB) return;
    int b = id >> 1;
    if ((id & 1) == 0) {
        float s = 0.f;
        for (int i = 0; i < NP / 256; ++i) s += psp[b * (NP / 256) + i];
        hacc[b * 2] = s;
    } else {
        float s = 0.f;
        for (int i = 0; i < NP / 128; ++i) s += pft[b * (NP / 128) + i];
        hacc[b * 2 + 1] = s;
    }
}

// ---------------------------------------------------------------------------
// K8: combine weights, update f, write point_features (d_out) and build the
// padded f16 activation matrix Xt[col][288] for the WMMA GEMM chain.
// ---------------------------------------------------------------------------
__global__ void fm_combine_kernel(const float* __restrict__ F, const float* __restrict__ pfeat,
                                  const int* __restrict__ idx, const float* __restrict__ dsp,
                                  const float* __restrict__ dft, const float* __restrict__ hacc,
                                  const float* __restrict__ brad,
                                  float* __restrict__ out_pf, _Float16* __restrict__ Xt) {
    int p = blockIdx.x * 128 + threadIdx.x;
    int b = p >> 12, n = p & (NP - 1);
    const float inv = 1.0f / (float)(NP * FMK);
    float hsp = hacc[b * 2] * inv;
    float hf  = hacc[b * 2 + 1] * inv;
    float asp = -2.0f / hsp;                   // exp(-d/(h/2)) = exp(-2d/h)
    float af  = -2.0f / hf;
    size_t base = (size_t)p * FMK;
    float w[FMK]; int mk[FMK]; float den = 0.f;
#pragma unroll
    for (int k = 0; k < FMK; ++k) {
        mk[k] = idx[base + k];
        w[k] = expf(asp * dsp[base + k]) * expf(af * dft[base + k]);
        den += w[k];
    }
    den += 5e-5f;                              // sum(w + 1e-5) over k=5
    float scl = 0.2f / den;
    const float* fq  = F + (size_t)p * CF;
    const float* pfb = pfeat + (size_t)b * CF * MP;
    float rad = brad[b];
    const float cx[RT] = {-0.2f, -0.2f, 0.2f, 0.2f};
    const float cy[RT] = {-0.2f, 0.2f, -0.2f, 0.2f};
    size_t colBase = (size_t)p * RT;
    float* pfo = out_pf + (size_t)b * CF * NP + n;
    for (int c = 0; c < CF; ++c) {
        float s = 0.f;
#pragma unroll
        for (int k = 0; k < FMK; ++k) s += w[k] * pfb[(size_t)c * MP + mk[k]];
        float v = fq[c] + scl * s;
        pfo[(size_t)c * NP] = v;               // point_features output
        _Float16 hv = (_Float16)v;
#pragma unroll
        for (int r = 0; r < RT; ++r) Xt[(colBase + r) * KP1 + c] = hv;
    }
#pragma unroll
    for (int r = 0; r < RT; ++r) {
        _Float16* xr = Xt + (colBase + r) * KP1;
        xr[264] = (_Float16)(rad * cx[r]);
        xr[265] = (_Float16)(rad * cy[r]);
#pragma unroll
        for (int c = 266; c < KP1; ++c) xr[c] = (_Float16)0.f;
    }
}

// ---------------------------------------------------------------------------
// K9: convert GEMM weights to padded f16 (row-major [out][Kpad])
// ---------------------------------------------------------------------------
__global__ void wprep_kernel(const float* __restrict__ w1, const float* __restrict__ w2,
                             const float* __restrict__ w3, const float* __restrict__ w4,
                             _Float16* __restrict__ p1, _Float16* __restrict__ p2,
                             _Float16* __restrict__ p3, _Float16* __restrict__ p4) {
    int id = blockIdx.x * 256 + threadIdx.x;
    if (id < 128 * KP1) {                                    // up1 128 x 288 (266 real)
        int r = id / KP1, c = id % KP1;
        p1[id] = (c < 266) ? (_Float16)w1[r * 266 + c] : (_Float16)0.f;
        return;
    }
    id -= 128 * KP1;
    if (id < 128 * 128) { p2[id] = (_Float16)w2[id]; return; }   // up2
    id -= 128 * 128;
    if (id < 64 * 128) { p3[id] = (_Float16)w3[id]; return; }    // fc1
    id -= 64 * 128;
    if (id < 16 * 64) {                                          // fc2 padded to 16 rows
        p4[id] = (id < 3 * 64) ? (_Float16)w4[id] : (_Float16)0.f;
    }
}

// ---------------------------------------------------------------------------
// WMMA fragment helpers — CDNA5 v_wmma_f32_16x16x32_f16 VGPR layouts
// A (16x32 f16): lane L -> M=L%16; VGPR j<4: K=(L<16?0:8)+2j ; VGPR j>=4: +16
// B (32x16 f16): lane L -> N=L%16; VGPR j<4: K=(L<16?0:8)+2j ; VGPR j>=4: +16
// ---------------------------------------------------------------------------
__device__ __forceinline__ v16h load_fragA(const _Float16* rowp, int k0, int lane) {
    union { v16h v; v2h p[8]; } u;
    const _Float16* rp = rowp + k0 + ((lane & 16) ? 8 : 0);
#pragma unroll
    for (int j = 0; j < 8; ++j) {
        int kk = ((j >> 2) << 4) + ((j & 3) << 1);
        u.p[j] = *(const v2h*)(rp + kk);
    }
    return u.v;
}
__device__ __forceinline__ v16h load_fragB(const _Float16* colp, int k0, int lane) {
    union { v16h v; v2h p[8]; } u;
    const _Float16* rp = colp + k0 + ((lane & 16) ? 8 : 0);
#pragma unroll
    for (int j = 0; j < 8; ++j) u.p[j] = *(const v2h*)(rp + ((j >> 2) << 4) + ((j & 3) << 1));
    return u.v;
}

// ---------------------------------------------------------------------------
// K10: WMMA GEMM  Y[col][row] = relu(W[row][:] . X[col][:] + bias[row])
// X: [COLS][Kp] f16, W: [Mp][Kp] f16, Y: [COLS][KpOut] f16
// Block = 256 threads = 8 waves; wave w owns row-tile w of one column tile.
// ---------------------------------------------------------------------------
__global__ void gemm_f16_kernel(const _Float16* __restrict__ X, const _Float16* __restrict__ W,
                                const float* __restrict__ bias, _Float16* __restrict__ Y,
                                int Kp, int KpOut, int M, int rowTiles, int doRelu) {
    int wave = threadIdx.x >> 5, lane = threadIdx.x & 31;
    if (wave >= rowTiles) return;
    int col0 = blockIdx.x * 16, row0 = wave * 16;
    const _Float16* wl = W + ((size_t)row0 + (lane & 15)) * Kp;
    const _Float16* xl = X + ((size_t)col0 + (lane & 15)) * Kp;
    v8f acc = {};
    for (int k0 = 0; k0 < Kp; k0 += 32) {
        v16h a = load_fragA(wl, k0, lane);
        v16h b = load_fragB(xl, k0, lane);
        acc = __builtin_amdgcn_wmma_f32_16x16x32_f16(false, a, false, b,
                                                     (short)0, acc, false, false);
    }
    int nloc = lane & 15, mbase = (lane & 16) ? 8 : 0;
    size_t cb = (size_t)(col0 + nloc) * KpOut;
#pragma unroll
    for (int v = 0; v < 8; ++v) {
        int row = row0 + mbase + v;
        float val = acc[v] + ((row < M) ? bias[row] : 0.f);
        if (doRelu) val = fmaxf(val, 0.f);
        Y[cb + row] = (_Float16)val;
    }
}

// ---------------------------------------------------------------------------
// K11: final WMMA GEMM (fc2, 3x64, padded to 16 rows) -> f32 x4 output
// x4 layout: (B, 3, N*RATIO) ; col = b*16384 + n*4 + r
// ---------------------------------------------------------------------------
__global__ void gemm_out_kernel(const _Float16* __restrict__ X, const _Float16* __restrict__ W,
                                const float* __restrict__ bias, float* __restrict__ out) {
    int wave = threadIdx.x >> 5, lane = threadIdx.x & 31;
    int colTile = blockIdx.x * 8 + wave;
    int col0 = colTile * 16;
    const _Float16* wl = W + (size_t)(lane & 15) * 64;
    const _Float16* xl = X + ((size_t)col0 + (lane & 15)) * 64;
    v8f acc = {};
    for (int k0 = 0; k0 < 64; k0 += 32) {
        v16h a = load_fragA(wl, k0, lane);
        v16h b = load_fragB(xl, k0, lane);
        acc = __builtin_amdgcn_wmma_f32_16x16x32_f16(false, a, false, b,
                                                     (short)0, acc, false, false);
    }
    int nloc = lane & 15, mbase = (lane & 16) ? 8 : 0;
    int col = col0 + nloc;
    int b = col >> 14, cbn = col & 16383;
#pragma unroll
    for (int v = 0; v < 8; ++v) {
        int row = mbase + v;
        if (row < 3) {
            float val = acc[v] + bias[row];
            out[((size_t)b * 3 + row) * (NP * RT) + cbn] = val;
        }
    }
}

// ---------------------------------------------------------------------------
// Host launcher
// ---------------------------------------------------------------------------
static inline size_t alignUp(size_t x) { return (x + 255) & ~(size_t)255; }

extern "C" void kernel_launch(void* const* d_in, const int* in_sizes, int n_in,
                              void* d_out, int out_size, void* d_ws, size_t ws_size,
                              hipStream_t stream) {
    (void)in_sizes; (void)n_in; (void)out_size; (void)ws_size;
    // jax pytree flatten order: xyz, batch_radius, previous_xyz, previous_feat,
    // then params dict sorted by key: d1,d2,d3,d4,fc1,fc2,l0,p2,p3,p4,up1,up2
    const float* xyz   = (const float*)d_in[0];
    const float* brad  = (const float*)d_in[1];
    const float* pxyz  = (const float*)d_in[2];
    const float* pfeat = (const float*)d_in[3];
    const float* dW[4][3], *dB[4][3];
    for (int s = 0; s < 4; ++s)
        for (int l = 0; l < 3; ++l) {
            dW[s][l] = (const float*)d_in[4 + s * 6 + l * 2];
            dB[s][l] = (const float*)d_in[4 + s * 6 + l * 2 + 1];
        }
    const float* fc1W = (const float*)d_in[28]; const float* fc1B = (const float*)d_in[29];
    const float* fc2W = (const float*)d_in[30]; const float* fc2B = (const float*)d_in[31];
    const float* l0W  = (const float*)d_in[32]; const float* l0B  = (const float*)d_in[33];
    const float* pW[3], *pB[3];                  // p2, p3, p4
    for (int s = 0; s < 3; ++s) { pW[s] = (const float*)d_in[34 + 2 * s]; pB[s] = (const float*)d_in[35 + 2 * s]; }
    const float* up1W = (const float*)d_in[40]; const float* up1B = (const float*)d_in[41];
    const float* up2W = (const float*)d_in[42]; const float* up2B = (const float*)d_in[43];

    // ---- workspace carve-up --------------------------------------------------
    char* base = (char*)d_ws;
    size_t off = 0;
    float* F      = (float*)(base + off); off += alignUp((size_t)BB * NP * CF * 4);
    float* feat24 = (float*)(base + off); off += alignUp((size_t)BB * NP * 24 * 4);
    float* stats  = (float*)(base + off); off += alignUp(BB * 4 * 4);
    float* hacc   = (float*)(base + off); off += alignUp(BB * 2 * 4);
    float* psp    = (float*)(base + off); off += alignUp((size_t)BB * (NP / 256) * 4);
    float* pft    = (float*)(base + off); off += alignUp((size_t)BB * (NP / 128) * 4);
    int*   knn    = (int*)(base + off);   off += alignUp((size_t)BB * NP * KNN_K * 4);
    int*   fmidx  = (int*)(base + off);   off += alignUp((size_t)BB * NP * FMK * 4);
    float* fmdsp  = (float*)(base + off); off += alignUp((size_t)BB * NP * FMK * 4);
    float* fmdft  = (float*)(base + off); off += alignUp((size_t)BB * NP * FMK * 4);
    _Float16* Wp1 = (_Float16*)(base + off); off += alignUp((size_t)128 * KP1 * 2);
    _Float16* Wp2 = (_Float16*)(base + off); off += alignUp((size_t)128 * 128 * 2);
    _Float16* Wp3 = (_Float16*)(base + off); off += alignUp((size_t)64 * 128 * 2);
    _Float16* Wp4 = (_Float16*)(base + off); off += alignUp((size_t)16 * 64 * 2);
    _Float16* Xt  = (_Float16*)(base + off); off += alignUp((size_t)COLS * KP1 * 2);
    _Float16* Y1  = (_Float16*)(base + off); off += alignUp((size_t)COLS * 128 * 2);
    _Float16* Y3  = (_Float16*)(base + off); off += alignUp((size_t)COLS * 64 * 2);
    _Float16* Y2  = Xt;                       // gemm2 reads Y1, may overwrite Xt

    float* out_x4 = (float*)d_out;                                  // (B,3,N*4)
    float* out_pf = (float*)d_out + (size_t)BB * 3 * NP * RT;       // (B,264,N)

    const int TP = BB * NP;                    // 16384 total points

    stats_kernel<<<BB, 256, 0, stream>>>(xyz, stats);
    l0_kernel<<<TP / 256, 256, 0, stream>>>(xyz, stats, l0W, l0B, feat24, F);

    const int chOff[4] = {180, 120, 60, 0};
    const int cin[3]   = {84, 144, 204};
    for (int s = 0; s < 4; ++s) {
        knn_wmma_kernel<<<BB * (NP / 256), 256, 0, stream>>>(feat24, knn);
        edge_kernel<<<TP / 128, 128, 0, stream>>>(feat24, knn,
                                                  dW[s][0], dB[s][0], dW[s][1], dB[s][1],
                                                  dW[s][2], dB[s][2], F, chOff[s]);
        if (s < 3)
            pconv_kernel<<<TP / 256, 256, 0, stream>>>(F, pW[s], pB[s], feat24,
                                                       chOff[s], cin[s]);
    }

    fm_knn_kernel<<<BB * (NP / 256), 256, 0, stream>>>(xyz, pxyz, fmidx, fmdsp, psp);
    fm_feat_kernel<<<TP / 128, 128, 0, stream>>>(F, pfeat, fmidx, fmdft, pft);
    fm_reduce_kernel<<<1, 32, 0, stream>>>(psp, pft, hacc);
    fm_combine_kernel<<<TP / 128, 128, 0, stream>>>(F, pfeat, fmidx, fmdsp, fmdft,
                                                    hacc, brad, out_pf, Xt);

    int wtot = 128 * KP1 + 128 * 128 + 64 * 128 + 16 * 64;
    wprep_kernel<<<(wtot + 255) / 256, 256, 0, stream>>>(up1W, up2W, fc1W, fc2W,
                                                         Wp1, Wp2, Wp3, Wp4);

    // GEMM chain: 266->128 (relu) -> 128 (relu) -> 64 (relu) -> 3
    gemm_f16_kernel<<<COLS / 16, 256, 0, stream>>>(Xt, Wp1, up1B, Y1, KP1, 128, 128, 8, 1);
    gemm_f16_kernel<<<COLS / 16, 256, 0, stream>>>(Y1, Wp2, up2B, Y2, 128, 128, 128, 8, 1);
    gemm_f16_kernel<<<COLS / 16, 256, 0, stream>>>(Y2, Wp3, fc1B, Y3, 128, 64, 64, 4, 1);
    gemm_out_kernel<<<COLS / (16 * 8), 256, 0, stream>>>(Y3, Wp4, fc2B, out_x4);
}
